// BandSplitModule_52029233823980
// MI455X (gfx1250) — matrix operand order repaired and checked by reading, not compiled, not run on hardware
//
#include <hip/hip_runtime.h>

typedef __attribute__((ext_vector_type(8)))  _Float16 v8h;
typedef __attribute__((ext_vector_type(16))) _Float16 v16h;
typedef __attribute__((ext_vector_type(8)))  float    v8f;
typedef __attribute__((ext_vector_type(4)))  float    v4f;

#define B_    8
#define C_    2
#define F_    1024
#define T_    1024
#define NB_   64
#define BW_   16
#define EMB_  128
#define FEAT_ 64
#define MT    64   // T rows per block

__launch_bounds__(256)
__global__ void bandsplit_wmma_kernel(const float* __restrict__ xr,
                                      const float* __restrict__ xi,
                                      const float* __restrict__ lnw,
                                      const float* __restrict__ lnb,
                                      const float* __restrict__ fcw,
                                      const float* __restrict__ fcb,
                                      float* __restrict__ out)
{
    __shared__ float As[MT][65];                       // fp32 staging, stride 65 -> conflict free
    __shared__ __align__(16) _Float16 A16[MT][72];     // f16 A, 144B rows (16B aligned)
    __shared__ __align__(16) _Float16 Bt[EMB_][72];    // f16 B^T [emb][feat]
    __shared__ float lnw_s[FEAT_], lnb_s[FEAT_], bias_s[EMB_];

    const int tid  = threadIdx.x;
    const int t0   = blockIdx.x * MT;
    const int band = blockIdx.y;
    const int b    = blockIdx.z;

    if (tid < FEAT_) {
        lnw_s[tid] = lnw[band * FEAT_ + tid];
        lnb_s[tid] = lnb[band * FEAT_ + tid];
    }
    if (tid < EMB_) bias_s[tid] = fcb[band * EMB_ + tid];

    // ---- stage x into As, float4 over t; 16 lanes per (c,reim,w) source row ----
    {
        const int tq   = tid & 15;   // float4 index along t
        const int rgrp = tid >> 4;   // 0..15
        #pragma unroll
        for (int it = 0; it < 4; ++it) {
            int sr   = it * 16 + rgrp;       // enumerates (c, reim, w)
            int c    = sr >> 5;
            int reim = (sr >> 4) & 1;
            int w    = sr & 15;
            const float* src = (reim ? xi : xr)
                + (((size_t)b * C_ + c) * F_ + (size_t)band * BW_ + w) * T_ + t0 + tq * 4;
            v4f v = *(const v4f*)src;
            int f = c * 32 + w * 2 + reim;   // view_as_real feature index
            #pragma unroll
            for (int j = 0; j < 4; ++j)
                As[tq * 4 + j][f] = v[j];
        }
    }

    // ---- stage fc_w[band] transposed as f16: Bt[e][k] ----
    {
        const float* wsrc = fcw + (size_t)band * FEAT_ * EMB_;
        #pragma unroll
        for (int it = 0; it < 32; ++it) {
            int idx = it * 256 + tid;        // coalesced over e
            int k = idx >> 7;
            int e = idx & 127;
            Bt[e][k] = (_Float16)wsrc[idx];
        }
    }
    __syncthreads();

    // ---- per-row LayerNorm (4 lanes per row), write f16 A ----
    {
        const int row = tid >> 2;
        const int q   = tid & 3;
        float vals[16];
        float s = 0.f, s2 = 0.f;
        #pragma unroll
        for (int j = 0; j < 16; ++j) {
            float v = As[row][q * 16 + j];
            vals[j] = v;
            s += v; s2 += v * v;
        }
        s  += __shfl_xor(s, 1);  s  += __shfl_xor(s, 2);
        s2 += __shfl_xor(s2, 1); s2 += __shfl_xor(s2, 2);
        float mean = s * (1.0f / 64.0f);
        float var  = s2 * (1.0f / 64.0f) - mean * mean;
        float rstd = rsqrtf(var + 1e-5f);
        v8h pack0 = {}, pack1 = {};
        #pragma unroll
        for (int j = 0; j < 8; ++j) {
            int f0 = q * 16 + j;
            int f1 = q * 16 + 8 + j;
            pack0[j] = (_Float16)((vals[j]     - mean) * rstd * lnw_s[f0] + lnb_s[f0]);
            pack1[j] = (_Float16)((vals[8 + j] - mean) * rstd * lnw_s[f1] + lnb_s[f1]);
        }
        *(v8h*)&A16[row][q * 16]     = pack0;
        *(v8h*)&A16[row][q * 16 + 8] = pack1;
    }
    __syncthreads();

    // ---- WMMA: wave -> one 16-row M tile x four 16-col N tiles ----
    const int wid   = tid >> 5;
    const int lane  = tid & 31;
    const int lrow  = lane & 15;
    const int lhalf = lane >> 4;
    const int mtile = wid & 3;      // 4 M tiles
    const int nhalf = wid >> 2;     // 2 wave groups along N

    // A fragment (16-bit A 16x32 layout): lane<16 holds K {0..7, 16..23}, lane>=16 holds K {8..15, 24..31}
    const _Float16* arow = &A16[mtile * 16 + lrow][0];
    v16h afrag[2];
    #pragma unroll
    for (int kh = 0; kh < 2; ++kh) {
        v8h lo = *(const v8h*)(arow + kh * 32 + lhalf * 8);
        v8h hi = *(const v8h*)(arow + kh * 32 + lhalf * 8 + 16);
        afrag[kh] = __builtin_shufflevector(lo, hi,
            0,1,2,3,4,5,6,7,8,9,10,11,12,13,14,15);
    }

    const size_t obase =
        (((size_t)b * NB_ + band) * T_ + t0 + mtile * 16 + lhalf * 8) * EMB_;

    #pragma unroll
    for (int j = 0; j < 4; ++j) {
        int n0 = (nhalf * 4 + j) * 16;
        // Bias depends only on N (= lane%16) -> splat into all 8 accumulator regs,
        // let WMMA accumulate it (removes epilogue adds + WMMA->VALU hazard NOPs).
        float bias = bias_s[n0 + lrow];
        v8f acc = { bias, bias, bias, bias, bias, bias, bias, bias };

        // B fragment (16-bit B 32x16): lane = column, lane<16 holds K {0..7, 16..23}, lane>=16 holds K {8..15, 24..31}
        const _Float16* brow = &Bt[n0 + lrow][0];
        #pragma unroll
        for (int kh = 0; kh < 2; ++kh) {
            v8h lo = *(const v8h*)(brow + kh * 32 + lhalf * 8);
            v8h hi = *(const v8h*)(brow + kh * 32 + lhalf * 8 + 16);
            v16h bfrag = __builtin_shufflevector(lo, hi,
                0,1,2,3,4,5,6,7,8,9,10,11,12,13,14,15);
            acc = __builtin_amdgcn_wmma_f32_16x16x32_f16(
                false, afrag[kh], false, bfrag, (short)0, acc, false, false);
        }
        // C/D layout: VGPR v -> M = v + lhalf*8, N = lane%16
        float* op = out + obase + n0 + lrow;
        #pragma unroll
        for (int v = 0; v < 8; ++v)
            op[(size_t)v * EMB_] = acc[v];
    }
}

extern "C" void kernel_launch(void* const* d_in, const int* in_sizes, int n_in,
                              void* d_out, int out_size, void* d_ws, size_t ws_size,
                              hipStream_t stream) {
    const float* xr  = (const float*)d_in[0];
    const float* xi  = (const float*)d_in[1];
    const float* lnw = (const float*)d_in[2];
    const float* lnb = (const float*)d_in[3];
    const float* fcw = (const float*)d_in[4];
    const float* fcb = (const float*)d_in[5];
    float* out = (float*)d_out;
    (void)in_sizes; (void)n_in; (void)out_size; (void)d_ws; (void)ws_size;

    dim3 grid(T_ / MT, NB_, B_);   // (16, 64, 8)
    bandsplit_wmma_kernel<<<grid, 256, 0, stream>>>(xr, xi, lnw, lnb, fcw, fcb, out);
}
